// CausalSelfAttention_53858889892361
// MI455X (gfx1250) — compile-verified, matrix operand order checked by reading, not run on hardware
//
#include <hip/hip_runtime.h>

typedef __attribute__((ext_vector_type(16))) __bf16 v16bf;
typedef __attribute__((ext_vector_type(8)))  __bf16 v8bf;
typedef __attribute__((ext_vector_type(4)))  __bf16 v4bf;
typedef __attribute__((ext_vector_type(2)))  __bf16 v2bf;
typedef __attribute__((ext_vector_type(8)))  float  v8f;

#define N_HEAD 12
#define HS     64
#define SEQ_T  1024
#define NBATCH 8
#define CDIM   768
#define C3DIM  2304
#define BLK_M  256          // rows per workgroup (8 waves x 32 rows)

// ---------- conversion helpers ----------

__device__ __forceinline__ v2bf cvt2bf(float lo, float hi) {
#if __has_builtin(__builtin_amdgcn_cvt_pk_bf16_f32)
  return __builtin_amdgcn_cvt_pk_bf16_f32(lo, hi);
#else
  v2bf r; r[0] = (__bf16)lo; r[1] = (__bf16)hi; return r;
#endif
}

__device__ __forceinline__ __bf16 f2bf(float f) {
#if __has_builtin(__builtin_amdgcn_cvt_pk_bf16_f32)
  v2bf t = __builtin_amdgcn_cvt_pk_bf16_f32(f, f);
  return t[0];
#else
  return (__bf16)f;
#endif
}

// Concatenate two b128 bf16 loads into one A/B fragment (register coalesce only)
__device__ __forceinline__ v16bf pack_bf16(const __bf16* p0, const __bf16* p1) {
  v8bf a = *(const v8bf*)p0;
  v8bf b = *(const v8bf*)p1;
  return __builtin_shufflevector(a, b, 0,1,2,3,4,5,6,7,8,9,10,11,12,13,14,15);
}

__device__ __forceinline__ v8f wmma_bf16(v16bf a, v16bf b, v8f c) {
  return __builtin_amdgcn_wmma_f32_16x16x32_bf16(false, a, false, b,
                                                 (short)0, c, false, false);
}

// CDNA5 async DMA: global -> LDS, 16B per lane, tracked by ASYNCcnt
__device__ __forceinline__ void async_b128_to_lds(const void* gptr, void* lptr) {
  unsigned lds = (unsigned)(unsigned long long)lptr;   // addr[31:0] = LDS offset
  unsigned long long ga = (unsigned long long)gptr;
  asm volatile("global_load_async_to_lds_b128 %0, %1, off"
               :: "v"(lds), "v"(ga) : "memory");
}
__device__ __forceinline__ void wait_async0() {
  asm volatile("s_wait_asynccnt 0x0" ::: "memory");
}

__device__ __forceinline__ float hmax16(float v) {
#pragma unroll
  for (int off = 1; off < 16; off <<= 1) v = fmaxf(v, __shfl_xor(v, off, 16));
  return v;
}
__device__ __forceinline__ float hsum16(float v) {
#pragma unroll
  for (int off = 1; off < 16; off <<= 1) v += __shfl_xor(v, off, 16);
  return v;
}

// ---------- kernel 0: one-pass fp32 -> bf16 (bandwidth bound, runs once) ----------

__global__ __launch_bounds__(256)
void cvt_kernel(const float* __restrict__ in, __bf16* __restrict__ out, int n8) {
  const int idx = blockIdx.x * 256 + threadIdx.x;
  if (idx >= n8) return;
  const float4 a = *(const float4*)(in + (size_t)idx * 8);
  const float4 b = *(const float4*)(in + (size_t)idx * 8 + 4);
  v2bf c0 = cvt2bf(a.x, a.y), c1 = cvt2bf(a.z, a.w);
  v2bf c2 = cvt2bf(b.x, b.y), c3 = cvt2bf(b.z, b.w);
  v4bf lo = __builtin_shufflevector(c0, c1, 0, 1, 2, 3);
  v4bf hi = __builtin_shufflevector(c2, c3, 0, 1, 2, 3);
  *(v8bf*)(out + (size_t)idx * 8) =
      __builtin_shufflevector(lo, hi, 0, 1, 2, 3, 4, 5, 6, 7);
}

// ---------- kernel 1: qkv = x @ W_attn^T + b_attn -> q/k/vT (bf16) ----------
// 256 threads = 8 waves; each wave owns a 32(M)x64(N) tile; the 64x32 B-chunk
// is shared via LDS, staged with async DMA and double-buffered.
// grid = (BT/256, 3C/64)

__global__ __launch_bounds__(256)
void qkv_kernel(const __bf16* __restrict__ xb, const __bf16* __restrict__ wb,
                const float* __restrict__ bias,
                __bf16* __restrict__ qw, __bf16* __restrict__ kw,
                __bf16* __restrict__ vtw) {
  __shared__ alignas(16) __bf16 tileB[2][64 * 32];
  const int tid  = threadIdx.x;
  const int lane = tid & 31;
  const int wid  = tid >> 5;
  const int l16  = lane & 15;
  const int hi   = (lane >> 4) & 1;
  const int Mb   = blockIdx.x * BLK_M + wid * 32;
  const int Nb   = blockIdx.y * 64;

  // this thread's 16B segment of the 64(row)x32(K) B chunk
  const int srow = tid >> 2;          // 0..63
  const int scol = (tid & 3) * 8;     // 0,8,16,24
  const __bf16* gsrc = wb + (size_t)(Nb + srow) * CDIM + scol;

  v8f acc[2][4] = {};
  const __bf16* xrow0 = xb + (size_t)(Mb + l16) * CDIM;
  const __bf16* xrow1 = xb + (size_t)(Mb + 16 + l16) * CDIM;

  async_b128_to_lds(gsrc, &tileB[0][srow * 32 + scol]);   // preload chunk 0

  int cur = 0;
  for (int kc = 0; kc < CDIM; kc += 32, cur ^= 1) {
    wait_async0();
    __syncthreads();                                      // buf[cur] ready everywhere
    if (kc + 32 < CDIM)
      async_b128_to_lds(gsrc + kc + 32, &tileB[cur ^ 1][srow * 32 + scol]);

    const int k0 = kc + hi * 8;
    v16bf a0 = pack_bf16(xrow0 + k0, xrow0 + k0 + 16);
    v16bf a1 = pack_bf16(xrow1 + k0, xrow1 + k0 + 16);
    __builtin_prefetch(xrow0 + k0 + 32, 0, 1);
    __builtin_prefetch(xrow1 + k0 + 32, 0, 1);
#pragma unroll
    for (int nb = 0; nb < 4; ++nb) {
      const __bf16* brow = &tileB[cur][(nb * 16 + l16) * 32 + hi * 8];
      v16bf bm = pack_bf16(brow, brow + 16);
      acc[0][nb] = wmma_bf16(a0, bm, acc[0][nb]);
      acc[1][nb] = wmma_bf16(a1, bm, acc[1][nb]);
    }
  }

#pragma unroll
  for (int mi = 0; mi < 2; ++mi) {
#pragma unroll
    for (int nb = 0; nb < 4; ++nb) {
      const int col = Nb + nb * 16 + l16;
      const float bv = bias[col];
#pragma unroll
      for (int i = 0; i < 8; ++i) {
        const int bt = Mb + mi * 16 + i + hi * 8;
        const int b  = bt >> 10;
        const int t  = bt & (SEQ_T - 1);
        const __bf16 o = f2bf(acc[mi][nb][i] + bv);
        if (col < CDIM) {                                    // Q: [B,H,T,hs]
          const int h = col >> 6, d = col & 63;
          qw[(((size_t)(b * N_HEAD + h)) * SEQ_T + t) * HS + d] = o;
        } else if (col < 2 * CDIM) {                         // K: [B,H,T,hs]
          const int c = col - CDIM, h = c >> 6, d = c & 63;
          kw[(((size_t)(b * N_HEAD + h)) * SEQ_T + t) * HS + d] = o;
        } else {                                             // V^T: [B,H,hs,T]
          const int c = col - 2 * CDIM, h = c >> 6, d = c & 63;
          vtw[(((size_t)(b * N_HEAD + h)) * HS + d) * SEQ_T + t] = o;
        }
      }
    }
  }
}

// ---------- kernel 2: flash causal attention, one wave per (b,h,query-tile) ----------
// grid = (T/16, B*H)

__global__ __launch_bounds__(32)
void attn_kernel(const __bf16* __restrict__ qw, const __bf16* __restrict__ kw,
                 const __bf16* __restrict__ vtw, __bf16* __restrict__ yw) {
  __shared__ alignas(16) __bf16 plds[16 * 40];   // 16x32 P tile (bf16), padded
  const int lane = threadIdx.x;
  const int l16  = lane & 15;
  const int hi   = lane >> 4;
  const int qt   = blockIdx.x;
  const int bh   = blockIdx.y;
  const int b    = bh / N_HEAD, h = bh % N_HEAD;

  const __bf16* qp = qw  + (size_t)bh * SEQ_T * HS;
  const __bf16* kp = kw  + (size_t)bh * SEQ_T * HS;
  const __bf16* vp = vtw + (size_t)bh * HS * SEQ_T;

  // Q fragments (d = 0..63 in two K-chunks), loaded once
  const int qrow = qt * 16 + l16;
  v16bf qa[2];
#pragma unroll
  for (int c = 0; c < 2; ++c) {
    const int d0 = 32 * c + hi * 8;
    qa[c] = pack_bf16(qp + qrow * HS + d0, qp + qrow * HS + d0 + 16);
  }

  v8f acc[4] = {};
  float rmax[8], rsum[8];
#pragma unroll
  for (int i = 0; i < 8; ++i) { rmax[i] = -3.0e38f; rsum[i] = 0.0f; }

  const int qEnd = qt * 16 + 15;
  for (int kc = 0; kc <= qEnd; kc += 32) {
    // --- scores: S = Q @ K^T for 32 keys (two 16-wide tiles) ---
    v8f s0 = {}, s1 = {};
#pragma unroll
    for (int c = 0; c < 2; ++c) {
      const int d0 = 32 * c + hi * 8;
      v16bf kb0 = pack_bf16(kp + (kc + l16) * HS + d0,
                            kp + (kc + l16) * HS + d0 + 16);
      s0 = wmma_bf16(qa[c], kb0, s0);
      v16bf kb1 = pack_bf16(kp + (kc + 16 + l16) * HS + d0,
                            kp + (kc + 16 + l16) * HS + d0 + 16);
      s1 = wmma_bf16(qa[c], kb1, s1);
    }

    // --- causal mask + online softmax update ---
    const float scale = 0.125f;                 // 1/sqrt(64)
    const int key0 = kc + l16, key1 = key0 + 16;
#pragma unroll
    for (int i = 0; i < 8; ++i) {
      const int qi = qt * 16 + i + hi * 8;
      const float a0 = (key0 <= qi) ? s0[i] * scale : -1.0e30f;
      const float a1 = (key1 <= qi) ? s1[i] * scale : -1.0e30f;
      const float cm = hmax16(fmaxf(a0, a1));   // row max over this chunk
      const float nm = fmaxf(rmax[i], cm);
      const float f  = __expf(rmax[i] - nm);
      rmax[i] = nm;
      const float e0 = __expf(a0 - nm);
      const float e1 = __expf(a1 - nm);
      rsum[i] = rsum[i] * f + hsum16(e0 + e1);
#pragma unroll
      for (int nb = 0; nb < 4; ++nb) acc[nb][i] *= f;
      // store P (bf16) in C-layout to LDS
      const int r = i + hi * 8;
      v2bf p = cvt2bf(e0, e1);
      plds[r * 40 + l16]      = p[0];
      plds[r * 40 + 16 + l16] = p[1];
    }
    __syncthreads();                            // single wave: DS ordering only

    // --- re-read P as A-layout fragment ---
    const int cb = hi * 8;
    v16bf pa = pack_bf16(&plds[l16 * 40 + cb], &plds[l16 * 40 + cb + 16]);
    __syncthreads();

    // --- accumulate O += P @ V (V^T rows are contiguous) ---
    const int kk0 = kc + hi * 8;
#pragma unroll
    for (int nb = 0; nb < 4; ++nb) {
      const int n = nb * 16 + l16;
      v16bf vb = pack_bf16(vp + n * SEQ_T + kk0, vp + n * SEQ_T + kk0 + 16);
      acc[nb] = wmma_bf16(pa, vb, acc[nb]);
    }
  }

  // --- normalize and write y in [B,T,C] bf16 for the projection ---
#pragma unroll
  for (int nb = 0; nb < 4; ++nb) {
    const int n = nb * 16 + l16;
#pragma unroll
    for (int i = 0; i < 8; ++i) {
      const int m = i + hi * 8;
      const int t = qt * 16 + m;
      yw[((size_t)(b * SEQ_T + t)) * CDIM + h * HS + n] = f2bf(acc[nb][i] / rsum[i]);
    }
  }
}

// ---------- kernel 3: out = y @ W_proj^T + b_proj (f32 out) ----------
// same 8-wave shared-B structure; grid = (BT/256, C/64)

__global__ __launch_bounds__(256)
void proj_kernel(const __bf16* __restrict__ yw, const __bf16* __restrict__ wb,
                 const float* __restrict__ bias, float* __restrict__ out) {
  __shared__ alignas(16) __bf16 tileB[2][64 * 32];
  const int tid  = threadIdx.x;
  const int lane = tid & 31;
  const int wid  = tid >> 5;
  const int l16  = lane & 15;
  const int hi   = (lane >> 4) & 1;
  const int Mb   = blockIdx.x * BLK_M + wid * 32;
  const int Nb   = blockIdx.y * 64;

  const int srow = tid >> 2;
  const int scol = (tid & 3) * 8;
  const __bf16* gsrc = wb + (size_t)(Nb + srow) * CDIM + scol;

  v8f acc[2][4] = {};
  const __bf16* yrow0 = yw + (size_t)(Mb + l16) * CDIM;
  const __bf16* yrow1 = yw + (size_t)(Mb + 16 + l16) * CDIM;

  async_b128_to_lds(gsrc, &tileB[0][srow * 32 + scol]);

  int cur = 0;
  for (int kc = 0; kc < CDIM; kc += 32, cur ^= 1) {
    wait_async0();
    __syncthreads();
    if (kc + 32 < CDIM)
      async_b128_to_lds(gsrc + kc + 32, &tileB[cur ^ 1][srow * 32 + scol]);

    const int k0 = kc + hi * 8;
    v16bf a0 = pack_bf16(yrow0 + k0, yrow0 + k0 + 16);
    v16bf a1 = pack_bf16(yrow1 + k0, yrow1 + k0 + 16);
    __builtin_prefetch(yrow0 + k0 + 32, 0, 1);
    __builtin_prefetch(yrow1 + k0 + 32, 0, 1);
#pragma unroll
    for (int nb = 0; nb < 4; ++nb) {
      const __bf16* brow = &tileB[cur][(nb * 16 + l16) * 32 + hi * 8];
      v16bf bm = pack_bf16(brow, brow + 16);
      acc[0][nb] = wmma_bf16(a0, bm, acc[0][nb]);
      acc[1][nb] = wmma_bf16(a1, bm, acc[1][nb]);
    }
  }

#pragma unroll
  for (int mi = 0; mi < 2; ++mi) {
#pragma unroll
    for (int nb = 0; nb < 4; ++nb) {
      const int col = Nb + nb * 16 + l16;
      const float bv = bias[col];
#pragma unroll
      for (int i = 0; i < 8; ++i) {
        const int m = mi * 16 + i + hi * 8;
        out[(size_t)(Mb + m) * CDIM + col] = acc[mi][nb][i] + bv;
      }
    }
  }
}

// ---------- launch ----------

extern "C" void kernel_launch(void* const* d_in, const int* in_sizes, int n_in,
                              void* d_out, int out_size, void* d_ws, size_t ws_size,
                              hipStream_t stream) {
  const float* x      = (const float*)d_in[0];
  const float* W_attn = (const float*)d_in[1];
  const float* b_attn = (const float*)d_in[2];
  const float* W_proj = (const float*)d_in[3];
  const float* b_proj = (const float*)d_in[4];
  float* out = (float*)d_out;

  const size_t QSZ  = (size_t)NBATCH * N_HEAD * SEQ_T * HS;  // 6 291 456
  const size_t XSZ  = (size_t)NBATCH * SEQ_T * CDIM;         // 6 291 456
  const size_t WASZ = (size_t)C3DIM * CDIM;                  // 1 769 472
  const size_t WPSZ = (size_t)CDIM * CDIM;                   //   589 824

  __bf16* qw   = (__bf16*)d_ws;
  __bf16* kw   = qw + QSZ;
  __bf16* vtw  = kw + QSZ;
  __bf16* yw   = vtw + QSZ;      // [B,T,C] bf16
  __bf16* xbf  = yw + XSZ;
  __bf16* wabf = xbf + XSZ;
  __bf16* wpbf = wabf + WASZ;

  // one-pass precision conversion (hoists all cvt out of GEMM hot loops)
  cvt_kernel<<<dim3((XSZ / 8 + 255) / 256), 256, 0, stream>>>(x, xbf, (int)(XSZ / 8));
  cvt_kernel<<<dim3((WASZ / 8 + 255) / 256), 256, 0, stream>>>(W_attn, wabf, (int)(WASZ / 8));
  cvt_kernel<<<dim3((WPSZ / 8 + 255) / 256), 256, 0, stream>>>(W_proj, wpbf, (int)(WPSZ / 8));

  const int BT = NBATCH * SEQ_T;                             // 8192 rows

  qkv_kernel<<<dim3(BT / BLK_M, C3DIM / 64), 256, 0, stream>>>(
      xbf, wabf, b_attn, qw, kw, vtw);

  attn_kernel<<<dim3(SEQ_T / 16, NBATCH * N_HEAD), 32, 0, stream>>>(
      qw, kw, vtw, yw);

  proj_kernel<<<dim3(BT / BLK_M, CDIM / 64), 256, 0, stream>>>(
      yw, wpbf, b_proj, out);
}
